// HierarchicalChunkFilter_14250701488176
// MI455X (gfx1250) — matrix operand order repaired and checked by reading, not compiled, not run on hardware
//
#include <hip/hip_runtime.h>

typedef __attribute__((ext_vector_type(16))) _Float16 v16h;
typedef __attribute__((ext_vector_type(8)))  float    v8f;
typedef _Float16 half_t;

#define BATCH   8
#define TLEN    2048
#define DIMIN   64
#define HIDD    256
#define CHUNKL  32
#define SAMPLES 32
#define NHEADS  4
#define DHEAD   64
#define FFDIM   512
#define SSTATE  8
#define OUTLEN  1024

// ---------------------------------------------------------------------------
// CDNA5 async global->LDS copy (GLOBAL_LOAD_ASYNC_TO_LDS_B128, ASYNCcnt).
// LDS byte address = flat shared pointer truncated to 32 bits (ISA: LDS
// aperture address is addr[31:0]).
// ---------------------------------------------------------------------------
__device__ inline void async_copy_b128(const void* gsrc, void* lds_dst) {
  const unsigned loff = (unsigned)(uintptr_t)lds_dst;
  const unsigned long long ga = (unsigned long long)(uintptr_t)gsrc;
  asm volatile("global_load_async_to_lds_b128 %0, %1, off" ::"v"(loff), "v"(ga)
               : "memory");
}
__device__ inline void wait_async_zero() {
  asm volatile("s_wait_asynccnt 0x0" ::: "memory");
}

// ---------------------------------------------------------------------------
// WMMA fragment helpers (wave32, V_WMMA_F32_16X16X32_F16)
// ---------------------------------------------------------------------------
union FragU {
  v16h     v;
  _Float16 h[16];
  unsigned u[8];
};

__device__ inline v8f wmma_f16(v16h a, v16h b, v8f c) {
  return __builtin_amdgcn_wmma_f32_16x16x32_f16(false, a, false, b, (short)0, c,
                                                false, false);
}

// A-matrix 16x32 f16 frag from row-major LDS tile. Lane L: m=L&15, half=L>>4.
// VGPR i holds K pair: i<4 -> 2i+8*half ; i>=4 -> 16+2(i-4)+8*half (ISA 7.12.2)
// All reads are contiguous-mergeable (2x ds_load_b128 per lane).
__device__ inline v16h load_a_frag(const half_t* lds, int row0, int stride, int k0) {
  const int lane = threadIdx.x & 31;
  const int m = lane & 15, half = lane >> 4;
  const half_t* rowp = lds + (size_t)(row0 + m) * stride + k0;
  FragU r;
#pragma unroll
  for (int i = 0; i < 8; ++i) {
    const int kp = (i < 4) ? (2 * i + 8 * half) : (16 + 2 * (i - 4) + 8 * half);
    r.u[i] = *(const unsigned*)(rowp + kp);
  }
  return r.v;
}

// B-frag where logical B[k][n] is stored transposed in LDS as row-major [N][K].
// Lane L: n=L&15; K pair i: i<4 -> 2i+8*half ; i>=4 -> 16+2(i-4)+8*half -> 16B reads that
// merge into ds_load_b128 (this replaces the strided element-gather path).
__device__ inline v16h load_bt_frag(const half_t* lds, int col0, int stride, int k0) {
  const int lane = threadIdx.x & 31;
  const int n = lane & 15, half = lane >> 4;
  const half_t* rowp = lds + (size_t)(col0 + n) * stride + k0 + 8 * half;
  FragU r;
#pragma unroll
  for (int i = 0; i < 8; ++i) r.u[i] = *(const unsigned*)(rowp + 2 * i + ((i >= 4) ? 8 : 0));
  return r.v;
}

// ---------------------------------------------------------------------------
// GEMM: C[M,N](f16) = act(A[M,K](f16) @ W[K,N](f32->f16) + bias)
// 128 threads = 4 waves; tile 64x64; each wave computes a 16x64 stripe.
// A tile staged global->LDS asynchronously (requires M % 64 == 0, true for
// every GEMM in this graph). W tile staged transposed with f32->f16 convert.
// ---------------------------------------------------------------------------
#define GBM 64
#define GBN 64
#define APAD 48    // 96B rows: 16B-aligned for b128 merging
#define BKPAD 48   // transposed B rows [N][K], 96B rows

__global__ __launch_bounds__(128) void gemm_bias_act_kernel(
    const half_t* __restrict__ A, const float* __restrict__ W,
    const float* __restrict__ bias, half_t* __restrict__ C, int M, int N, int K,
    int relu) {
  __shared__ half_t As[GBM * APAD];
  __shared__ half_t Bs[GBN * BKPAD];
  const int tid = threadIdx.x;
  const int wave = tid >> 5;
  const int tile_m = blockIdx.x * GBM;
  const int tile_n = blockIdx.y * GBN;

  v8f acc[4];
#pragma unroll
  for (int n = 0; n < 4; ++n)
#pragma unroll
    for (int r = 0; r < 8; ++r) acc[n][r] = 0.0f;

  const int arow = tid >> 1, aseg = (tid & 1) * 16;
  const int brow = tid >> 2, bcseg = (tid & 3) * 16;

  for (int k0 = 0; k0 < K; k0 += 32) {
    __syncthreads();
    {  // async stage A tile 64x32 f16 (2 threads/row, 2x b128 each)
      const half_t* ga = A + (size_t)(tile_m + arow) * K + k0 + aseg;
      half_t* la = As + (size_t)arow * APAD + aseg;
      async_copy_b128(ga, la);
      async_copy_b128(ga + 8, la + 8);
    }
    {  // stage W tile 32x64 transposed: Bs[n][k] = (f16)W[k0+k][tile_n+n]
      const int gc = tile_n + bcseg;
      float tmp[16];
      if (gc < N) {
        const float4* src = (const float4*)(W + (size_t)(k0 + brow) * N + gc);
#pragma unroll
        for (int q = 0; q < 4; ++q) {
          float4 f = src[q];
          tmp[q * 4 + 0] = f.x; tmp[q * 4 + 1] = f.y;
          tmp[q * 4 + 2] = f.z; tmp[q * 4 + 3] = f.w;
        }
      } else {
#pragma unroll
        for (int q = 0; q < 16; ++q) tmp[q] = 0.0f;
      }
#pragma unroll
      for (int j = 0; j < 16; ++j)
        Bs[(size_t)(bcseg + j) * BKPAD + brow] = (half_t)tmp[j];
    }
    // warm GL2/L0 for next weight slice (global_prefetch_b8)
    if (k0 + 32 < K && tile_n + bcseg < N)
      __builtin_prefetch(W + (size_t)(k0 + 32 + brow) * N + tile_n + bcseg, 0, 0);
    wait_async_zero();
    __syncthreads();
    const v16h a = load_a_frag(As, wave * 16, APAD, 0);
#pragma unroll
    for (int n = 0; n < 4; ++n) {
      const v16h b = load_bt_frag(Bs, n * 16, BKPAD, 0);
      acc[n] = wmma_f16(a, b, acc[n]);
    }
  }

  const int lane = tid & 31;
  const int nn = lane & 15, hs = lane >> 4;
#pragma unroll
  for (int n = 0; n < 4; ++n) {
    const int gc = tile_n + n * 16 + nn;
    if (gc >= N) continue;
    const float bv = bias ? bias[gc] : 0.0f;
#pragma unroll
    for (int r = 0; r < 8; ++r) {
      const int gm = tile_m + wave * 16 + r + 8 * hs;
      if (gm >= M) continue;
      float v = acc[n][r] + bv;
      if (relu) v = fmaxf(v, 0.0f);
      C[(size_t)gm * N + gc] = (half_t)v;
    }
  }
}

// ---------------------------------------------------------------------------
// Embedding: emb = (data[M,64] @ We[64,256] + be + PE[t]) * sqrt(32)
// A staged from f32 (needs conversion, so synchronous); W transposed like GEMM.
// ---------------------------------------------------------------------------
__device__ inline float pe_val(int t, int n) {
  const int j = n >> 1;
  const float f = __expf(-(float)(2 * j) * (9.210340371976184f / 256.0f));
  const float a = (float)t * f;
  return (n & 1) ? __cosf(a) : __sinf(a);
}

__global__ __launch_bounds__(128) void embed_kernel(
    const float* __restrict__ data, const float* __restrict__ W,
    const float* __restrict__ bias, half_t* __restrict__ emb16,
    float* __restrict__ emb32, int M) {
  __shared__ half_t As[GBM * APAD];
  __shared__ half_t Bs[GBN * BKPAD];
  const int tid = threadIdx.x;
  const int wave = tid >> 5;
  const int tile_m = blockIdx.x * GBM;
  const int tile_n = blockIdx.y * GBN;
  const int K = DIMIN, N = HIDD;

  v8f acc[4];
#pragma unroll
  for (int n = 0; n < 4; ++n)
#pragma unroll
    for (int r = 0; r < 8; ++r) acc[n][r] = 0.0f;

  for (int k0 = 0; k0 < K; k0 += 32) {
    __syncthreads();
    {  // stage A from f32
      const int row = tid >> 1;
      const int seg = (tid & 1) * 16;
      const int gr = tile_m + row;
      const float4* src = (const float4*)(data + (size_t)gr * K + k0 + seg);
#pragma unroll
      for (int q = 0; q < 4; ++q) {
        float4 f = src[q];
        As[(size_t)row * APAD + seg + q * 4 + 0] = (half_t)f.x;
        As[(size_t)row * APAD + seg + q * 4 + 1] = (half_t)f.y;
        As[(size_t)row * APAD + seg + q * 4 + 2] = (half_t)f.z;
        As[(size_t)row * APAD + seg + q * 4 + 3] = (half_t)f.w;
      }
    }
    {  // stage W transposed
      const int row = tid >> 2;
      const int cseg = (tid & 3) * 16;
      const int gc = tile_n + cseg;
      const float4* src = (const float4*)(W + (size_t)(k0 + row) * N + gc);
#pragma unroll
      for (int q = 0; q < 4; ++q) {
        float4 f = src[q];
        Bs[(size_t)(cseg + q * 4 + 0) * BKPAD + row] = (half_t)f.x;
        Bs[(size_t)(cseg + q * 4 + 1) * BKPAD + row] = (half_t)f.y;
        Bs[(size_t)(cseg + q * 4 + 2) * BKPAD + row] = (half_t)f.z;
        Bs[(size_t)(cseg + q * 4 + 3) * BKPAD + row] = (half_t)f.w;
      }
    }
    __syncthreads();
    const v16h a = load_a_frag(As, wave * 16, APAD, 0);
#pragma unroll
    for (int n = 0; n < 4; ++n) {
      const v16h b = load_bt_frag(Bs, n * 16, BKPAD, 0);
      acc[n] = wmma_f16(a, b, acc[n]);
    }
  }

  const int lane = tid & 31;
  const int nn = lane & 15, hs = lane >> 4;
#pragma unroll
  for (int n = 0; n < 4; ++n) {
    const int gc = tile_n + n * 16 + nn;
    const float bv = bias[gc];
#pragma unroll
    for (int r = 0; r < 8; ++r) {
      const int gm = tile_m + wave * 16 + r + 8 * hs;
      if (gm >= M) continue;
      const int t = gm & (TLEN - 1);
      const float v = (acc[n][r] + bv + pe_val(t, gc)) * 5.656854249492381f;
      emb16[(size_t)gm * HIDD + gc] = (half_t)v;
      emb32[(size_t)gm * HIDD + gc] = v;
    }
  }
}

// ---------------------------------------------------------------------------
// Attention: one workgroup per (sequence, head). Lq,Lk <= 64, dh = 64.
// Q/K tiles staged via async global->LDS (valid rows) + zero fill (pad rows);
// V staged transposed so the P@V B-fragments are contiguous b128 LDS loads.
// ---------------------------------------------------------------------------
#define ATPAD 72
#define SSPAD 68

__global__ __launch_bounds__(128) void attn_kernel(
    const half_t* __restrict__ Q, const half_t* __restrict__ K,
    const half_t* __restrict__ V, half_t* __restrict__ O, int Lq, int Lk,
    int kvg) {
  __shared__ half_t Qs[64 * ATPAD];
  __shared__ half_t Ks[64 * ATPAD];
  __shared__ half_t Vt[64 * ATPAD];  // Vt[d][k] = V[k][d]
  __shared__ half_t Ps[64 * ATPAD];
  __shared__ float  Ss[64 * SSPAD];

  const int seq = blockIdx.x;
  const int h = blockIdx.y;
  const int tid = threadIdx.x;
  const int wave = tid >> 5;
  const int lane = tid & 31;

  const half_t* qp = Q + ((size_t)seq * Lq) * HIDD + h * DHEAD;
  const half_t* kp = K + ((size_t)(seq / kvg) * Lk) * HIDD + h * DHEAD;
  const half_t* vp = V + ((size_t)(seq / kvg) * Lk) * HIDD + h * DHEAD;

  {  // stage: 2 threads/row, 32 f16 (64B) per thread per matrix
    const int row = tid >> 1;
    const int cseg = (tid & 1) * 32;
    const uint4 z = {0u, 0u, 0u, 0u};
    if (row < Lq) {
      const half_t* g = qp + (size_t)row * HIDD + cseg;
      half_t* l = Qs + (size_t)row * ATPAD + cseg;
      async_copy_b128(g, l);
      async_copy_b128(g + 8, l + 8);
      async_copy_b128(g + 16, l + 16);
      async_copy_b128(g + 24, l + 24);
    } else {
      uint4* l = (uint4*)(Qs + (size_t)row * ATPAD + cseg);
      l[0] = z; l[1] = z; l[2] = z; l[3] = z;
    }
    if (row < Lk) {
      const half_t* g = kp + (size_t)row * HIDD + cseg;
      half_t* l = Ks + (size_t)row * ATPAD + cseg;
      async_copy_b128(g, l);
      async_copy_b128(g + 8, l + 8);
      async_copy_b128(g + 16, l + 16);
      async_copy_b128(g + 24, l + 24);
    } else {
      uint4* l = (uint4*)(Ks + (size_t)row * ATPAD + cseg);
      l[0] = z; l[1] = z; l[2] = z; l[3] = z;
    }
    // V: read row, write transposed
    union { uint4 q[4]; _Float16 hh[32]; } vb;
    vb.q[0] = z; vb.q[1] = z; vb.q[2] = z; vb.q[3] = z;
    if (row < Lk) {
      const uint4* s = (const uint4*)(vp + (size_t)row * HIDD + cseg);
      vb.q[0] = s[0]; vb.q[1] = s[1]; vb.q[2] = s[2]; vb.q[3] = s[3];
    }
#pragma unroll
    for (int j = 0; j < 32; ++j)
      Vt[(size_t)(cseg + j) * ATPAD + row] = vb.hh[j];
  }
  wait_async_zero();
  __syncthreads();

  // Phase 1: S = Q K^T * (1/sqrt(64));  B = K^T read via transposed frag
  v8f sacc[4];
#pragma unroll
  for (int n = 0; n < 4; ++n)
#pragma unroll
    for (int r = 0; r < 8; ++r) sacc[n][r] = 0.0f;
#pragma unroll
  for (int kt = 0; kt < 64; kt += 32) {
    const v16h a = load_a_frag(Qs, wave * 16, ATPAD, kt);
#pragma unroll
    for (int nt = 0; nt < 4; ++nt) {
      const v16h b = load_bt_frag(Ks, nt * 16, ATPAD, kt);
      sacc[nt] = wmma_f16(a, b, sacc[nt]);
    }
  }
  {
    const int nn = lane & 15, hs = lane >> 4;
#pragma unroll
    for (int nt = 0; nt < 4; ++nt)
#pragma unroll
      for (int r = 0; r < 8; ++r)
        Ss[(size_t)(wave * 16 + r + 8 * hs) * SSPAD + nt * 16 + nn] =
            sacc[nt][r] * 0.125f;
  }
  __syncthreads();

  // f32 softmax over valid columns; P padded with zeros beyond Lk
  if (tid < 64) {
    float* srow = Ss + (size_t)tid * SSPAD;
    float mx = -1e30f;
    for (int c = 0; c < Lk; ++c) mx = fmaxf(mx, srow[c]);
    float sum = 0.0f;
    for (int c = 0; c < Lk; ++c) {
      const float e = __expf(srow[c] - mx);
      srow[c] = e;
      sum += e;
    }
    const float inv = 1.0f / sum;
    half_t* prow = Ps + (size_t)tid * ATPAD;
    for (int c = 0; c < 64; ++c)
      prow[c] = (c < Lk) ? (half_t)(srow[c] * inv) : (half_t)0.0f;
  }
  __syncthreads();

  // Phase 2: O = P V;  B = V read via transposed frag from Vt
  v8f oacc[4];
#pragma unroll
  for (int n = 0; n < 4; ++n)
#pragma unroll
    for (int r = 0; r < 8; ++r) oacc[n][r] = 0.0f;
#pragma unroll
  for (int kt = 0; kt < 64; kt += 32) {
    const v16h a = load_a_frag(Ps, wave * 16, ATPAD, kt);
#pragma unroll
    for (int nt = 0; nt < 4; ++nt) {
      const v16h b = load_bt_frag(Vt, nt * 16, ATPAD, kt);
      oacc[nt] = wmma_f16(a, b, oacc[nt]);
    }
  }
  {
    const int nn = lane & 15, hs = lane >> 4;
#pragma unroll
    for (int nt = 0; nt < 4; ++nt)
#pragma unroll
      for (int r = 0; r < 8; ++r) {
        const int m = wave * 16 + r + 8 * hs;
        if (m < Lq)
          O[((size_t)seq * Lq + m) * HIDD + h * DHEAD + nt * 16 + nn] =
              (half_t)oacc[nt][r];
      }
  }
}

// ---------------------------------------------------------------------------
// LayerNorm(X [+ Y]) * g + b, one wave (32 lanes) per 256-wide row
// ---------------------------------------------------------------------------
__global__ __launch_bounds__(256) void ln_kernel(
    const half_t* __restrict__ X, const half_t* __restrict__ Yres,
    const float* __restrict__ g, const float* __restrict__ beta,
    half_t* __restrict__ Out, int M) {
  const int row = blockIdx.x * 8 + (threadIdx.x >> 5);
  if (row >= M) return;
  const int lane = threadIdx.x & 31;
  const half_t* xr = X + (size_t)row * HIDD;
  const half_t* yr = Yres ? Yres + (size_t)row * HIDD : nullptr;
  float v[8];
  float s = 0.0f;
#pragma unroll
  for (int i = 0; i < 8; ++i) {
    const int c = i * 32 + lane;
    float t = (float)xr[c];
    if (yr) t += (float)yr[c];
    v[i] = t;
    s += t;
  }
#pragma unroll
  for (int off = 16; off; off >>= 1) s += __shfl_xor(s, off, 32);
  const float mean = s * (1.0f / 256.0f);
  float vs = 0.0f;
#pragma unroll
  for (int i = 0; i < 8; ++i) {
    const float d = v[i] - mean;
    vs += d * d;
  }
#pragma unroll
  for (int off = 16; off; off >>= 1) vs += __shfl_xor(vs, off, 32);
  const float inv = rsqrtf(vs * (1.0f / 256.0f) + 1e-5f);
  half_t* orow = Out + (size_t)row * HIDD;
#pragma unroll
  for (int i = 0; i < 8; ++i) {
    const int c = i * 32 + lane;
    orow[c] = (half_t)((v[i] - mean) * inv * g[c] + beta[c]);
  }
}

// ---------------------------------------------------------------------------
// Small utility kernels
// ---------------------------------------------------------------------------
__global__ void cvt_f32_f16_kernel(const float* __restrict__ s,
                                   half_t* __restrict__ d, int n) {
  const int i = blockIdx.x * 256 + threadIdx.x;
  if (i < n) d[i] = (half_t)s[i];
}

__global__ void zero_f32_kernel(float* __restrict__ p, int n) {
  const int i = blockIdx.x * 256 + threadIdx.x;
  if (i < n) p[i] = 0.0f;
}

// dst row r <- src row r*rstride (res[lev] = x[:, :, 0] extraction)
__global__ void gather_stride_kernel(const half_t* __restrict__ src,
                                     half_t* __restrict__ dst, int rstride) {
  const int r = blockIdx.x, c = threadIdx.x;
  dst[(size_t)r * HIDD + c] = src[((size_t)r * rstride) * HIDD + c];
}

// keep first 32 of every 64 rows (comp[:, :CHUNK])
__global__ void compact_half_kernel(const half_t* __restrict__ src,
                                    half_t* __restrict__ dst) {
  const int g = blockIdx.x, r = blockIdx.y, c = threadIdx.x;
  dst[((size_t)g * 32 + r) * HIDD + c] = src[((size_t)g * 64 + r) * HIDD + c];
}

// dst[b*S+s] = src[b*Tl + idx[b*S+s]]  (one-hot einsum == gather)
__global__ void gather_idx_kernel(const half_t* __restrict__ src,
                                  const int* __restrict__ idx,
                                  half_t* __restrict__ dst, int S, int Tl) {
  const int i = blockIdx.x, c = threadIdx.x;
  const int b = i / S;
  dst[(size_t)i * HIDD + c] = src[((size_t)b * Tl + idx[i]) * HIDD + c];
}

__global__ void concat_kernel(const half_t* __restrict__ a,
                              const half_t* __restrict__ b,
                              half_t* __restrict__ dst) {
  const int r = blockIdx.x, c = threadIdx.x;
  dst[(size_t)r * 512 + c] = a[(size_t)r * HIDD + c];
  dst[(size_t)r * 512 + 256 + c] = b[(size_t)r * HIDD + c];
}

__device__ inline unsigned hashu(unsigned x) {
  x ^= x >> 16; x *= 0x7feb352dU; x ^= x >> 15; x *= 0x846ca68bU; x ^= x >> 16;
  return x;
}
__device__ inline float urand3(unsigned a, unsigned b, unsigned c) {
  const unsigned x =
      hashu(a * 0x9e3779b9U + hashu(b * 0x85ebca6bU + hashu(c + 0x27d4eb2fU)));
  return (float)(x >> 8) * (1.0f / 16777216.0f);
}

// categorical over Ct rows of logits[b*Ct+t][s]
__global__ void sample_top_kernel(const half_t* __restrict__ logits,
                                  int* __restrict__ outIdx, int Ct, int S,
                                  unsigned seed) {
  const int i = blockIdx.x * blockDim.x + threadIdx.x;
  if (i >= BATCH * S) return;
  const int b = i / S, s = i % S;
  float mx = -1e30f;
  for (int t = 0; t < Ct; ++t)
    mx = fmaxf(mx, (float)logits[((size_t)b * Ct + t) * S + s]);
  float sum = 0.0f;
  for (int t = 0; t < Ct; ++t)
    sum += __expf((float)logits[((size_t)b * Ct + t) * S + s] - mx);
  const float u = urand3((unsigned)b, (unsigned)s, seed) * sum;
  int pick = Ct - 1;
  float cdf = 0.0f;
  for (int t = 0; t < Ct; ++t) {
    cdf += __expf((float)logits[((size_t)b * Ct + t) * S + s] - mx);
    if (u < cdf) { pick = t; break; }
  }
  outIdx[i] = pick;
}

// sample among the two children {2p, 2p+1} of parent idx (masked _st_sample)
__global__ void sample_child_kernel(const half_t* __restrict__ logits,
                                    const int* __restrict__ pidx,
                                    int* __restrict__ outIdx, int Tl, int S,
                                    unsigned seed) {
  const int i = blockIdx.x * blockDim.x + threadIdx.x;
  if (i >= BATCH * S) return;
  const int b = i / S, s = i % S;
  const int t0 = 2 * pidx[i], t1 = t0 + 1;
  const float l0 = (float)logits[((size_t)b * Tl + t0) * S + s];
  const float l1 = (float)logits[((size_t)b * Tl + t1) * S + s];
  const float mx = fmaxf(l0, l1);
  const float e0 = __expf(l0 - mx), e1 = __expf(l1 - mx);
  const float u = urand3((unsigned)b, (unsigned)(s + 131), seed) * (e0 + e1);
  outIdx[i] = (u < e0) ? t0 : t1;
}

// per-b chunk counts (F.sum over samples) and stable ranks of selected chunks
__global__ void count_rank_kernel(const int* __restrict__ idx0,
                                  int* __restrict__ cnt,
                                  int* __restrict__ rank) {
  __shared__ int lc[64];
  const int b = blockIdx.x;
  if (threadIdx.x < 64) lc[threadIdx.x] = 0;
  __syncthreads();
  if (threadIdx.x < SAMPLES) atomicAdd(&lc[idx0[b * SAMPLES + threadIdx.x]], 1);
  __syncthreads();
  if (threadIdx.x == 0) {
    int r = 0;
    for (int c = 0; c < 64; ++c) {
      rank[b * 64 + c] = r;
      if (lc[c] > 0) r++;
    }
  }
  __syncthreads();
  if (threadIdx.x < 64) cnt[b * 64 + threadIdx.x] = lc[threadIdx.x];
}

// out[b, rank*32 + r, :] = emb32[b, c*32 + r, :] * cnt   for selected chunks
__global__ void scatter_out_kernel(const float* __restrict__ emb32,
                                   const int* __restrict__ cnt,
                                   const int* __restrict__ rank,
                                   float* __restrict__ out) {
  const int c = blockIdx.x, b = blockIdx.y, col = threadIdx.x;
  const int k = cnt[b * 64 + c];
  if (k == 0) return;
  const int rk = rank[b * 64 + c];
  const float f = (float)k;
  for (int r = 0; r < 32; ++r)
    out[((size_t)b * OUTLEN + rk * 32 + r) * HIDD + col] =
        emb32[((size_t)b * TLEN + c * 32 + r) * HIDD + col] * f;
}

// ---------------------------------------------------------------------------
// Host orchestration
// ---------------------------------------------------------------------------
extern "C" void kernel_launch(void* const* d_in, const int* in_sizes, int n_in,
                              void* d_out, int out_size, void* d_ws,
                              size_t ws_size, hipStream_t stream) {
  (void)in_sizes; (void)n_in; (void)out_size; (void)ws_size;

  // ---- parameter walk (setup_inputs insertion order, recursively) ----
  struct Lin { const float* w; const float* b; };
  struct LNp { const float* g; const float* b; };
  struct MHA { Lin q, k, v, o; };
  struct EncL { MHA sa; LNp ln1; Lin ff1, ff2; LNp ln2; };
  struct DecL { MHA sa; LNp ln1; MHA ca; LNp ln2; Lin ff1, ff2; LNp ln3; };
  struct TrP { EncL enc[2]; DecL dec[2]; int ne, nd; LNp en, dn; };

  int gi = 0;
  auto nfp = [&]() { return (const float*)d_in[gi++]; };
  auto nLin = [&]() { Lin l; l.w = nfp(); l.b = nfp(); return l; };
  auto nLN = [&]() { LNp l; l.g = nfp(); l.b = nfp(); return l; };
  auto nMHA = [&]() { MHA m; m.q = nLin(); m.k = nLin(); m.v = nLin(); m.o = nLin(); return m; };
  auto nEnc = [&]() { EncL e; e.sa = nMHA(); e.ln1 = nLN(); e.ff1 = nLin(); e.ff2 = nLin(); e.ln2 = nLN(); return e; };
  auto nDec = [&]() { DecL d; d.sa = nMHA(); d.ln1 = nLN(); d.ca = nMHA(); d.ln2 = nLN(); d.ff1 = nLin(); d.ff2 = nLin(); d.ln3 = nLN(); return d; };
  auto nTr = [&](int ne, int nd) {
    TrP t; t.ne = ne; t.nd = nd;
    for (int i = 0; i < ne; ++i) t.enc[i] = nEnc();
    for (int i = 0; i < nd; ++i) t.dec[i] = nDec();
    t.en = nLN(); t.dn = nLN(); return t;
  };

  const float* stateF = nfp();  // [B, 8, 256]
  const float* dataF = nfp();   // [B, 2048, 64]
  Lin embed = nLin();
  TrP levels[3];
  for (int l = 0; l < 3; ++l) levels[l] = nTr(1, 1);
  Lin head = nLin();
  Lin hbl1[3], hbl2[3];
  for (int l = 0; l < 3; ++l) { hbl1[l] = nLin(); hbl2[l] = nLin(); }
  Lin rnn1 = nLin(), rnn2 = nLin();
  TrP mixin = nTr(2, 2);

  // ---- workspace arena ----
  char* base = (char*)d_ws;
  size_t cur = 0;
  auto alloc = [&](size_t bytes) {
    void* p = base + cur;
    cur += (bytes + 255) & ~(size_t)255;
    return p;
  };
  const int NTOK = BATCH * TLEN;  // 16384
  half_t* emb16 = (half_t*)alloc((size_t)NTOK * HIDD * 2);
  float* emb32 = (float*)alloc((size_t)NTOK * HIDD * 4);
  half_t* X = (half_t*)alloc((size_t)NTOK * HIDD * 2);
  half_t* bQ = (half_t*)alloc((size_t)NTOK * HIDD * 2);
  half_t* bK = (half_t*)alloc((size_t)NTOK * HIDD * 2);
  half_t* bV = (half_t*)alloc((size_t)NTOK * HIDD * 2);
  half_t* bA = (half_t*)alloc((size_t)NTOK * HIDD * 2);
  half_t* bY = (half_t*)alloc((size_t)NTOK * HIDD * 2);
  half_t* bF = (half_t*)alloc((size_t)NTOK * FFDIM * 2);
  half_t* Xc = (half_t*)alloc((size_t)8192 * HIDD * 2);
  half_t* Xc2 = (half_t*)alloc((size_t)4096 * HIDD * 2);
  half_t* state16 = (half_t*)alloc((size_t)BATCH * SSTATE * HIDD * 2);
  half_t* memBuf = (half_t*)alloc((size_t)BATCH * 32 * HIDD * 2);
  half_t* res0 = (half_t*)alloc((size_t)512 * HIDD * 2);
  half_t* res1 = (half_t*)alloc((size_t)256 * HIDD * 2);
  half_t* res2 = (half_t*)alloc((size_t)128 * HIDD * 2);
  half_t* Fdata = (half_t*)alloc((size_t)256 * HIDD * 2);
  half_t* Fd1 = (half_t*)alloc((size_t)256 * HIDD * 2);
  half_t* catB = (half_t*)alloc((size_t)256 * 512 * 2);
  half_t* Xm = (half_t*)alloc((size_t)512 * HIDD * 2);
  half_t* Lh = (half_t*)alloc((size_t)512 * SAMPLES * 2);
  int* idx2 = (int*)alloc(256 * 4);
  int* idx1 = (int*)alloc(256 * 4);
  int* idx0 = (int*)alloc(256 * 4);
  int* cnt = (int*)alloc(BATCH * 64 * 4);
  int* rank = (int*)alloc(BATCH * 64 * 4);

  // ---- launch helpers ----
  auto gemm = [&](const half_t* A, Lin L, half_t* C, int M, int N, int K,
                  int relu) {
    dim3 grid((M + GBM - 1) / GBM, (N + GBN - 1) / GBN);
    gemm_bias_act_kernel<<<grid, 128, 0, stream>>>(A, L.w, L.b, C, M, N, K, relu);
  };
  auto attn = [&](const half_t* Q, const half_t* K, const half_t* V, half_t* O,
                  int Lq, int Lk, int kvg, int nseq) {
    attn_kernel<<<dim3(nseq, NHEADS), 128, 0, stream>>>(Q, K, V, O, Lq, Lk, kvg);
  };
  auto lnorm = [&](const half_t* Xp, const half_t* Yp, LNp p, half_t* Op, int M) {
    ln_kernel<<<(M + 7) / 8, 256, 0, stream>>>(Xp, Yp, p.g, p.b, Op, M);
  };

  auto run_encoder = [&](const TrP& tp, const half_t* src, half_t* m, int nseq,
                         int Ls) {
    const int ntok = nseq * Ls;
    hipMemcpyAsync(m, src, (size_t)ntok * HIDD * 2, hipMemcpyDeviceToDevice, stream);
    for (int i = 0; i < tp.ne; ++i) {
      const EncL& L = tp.enc[i];
      gemm(m, L.sa.q, bQ, ntok, HIDD, HIDD, 0);
      gemm(m, L.sa.k, bK, ntok, HIDD, HIDD, 0);
      gemm(m, L.sa.v, bV, ntok, HIDD, HIDD, 0);
      attn(bQ, bK, bV, bA, Ls, Ls, 1, nseq);
      gemm(bA, L.sa.o, bY, ntok, HIDD, HIDD, 0);
      lnorm(m, bY, L.ln1, m, ntok);
      gemm(m, L.ff1, bF, ntok, FFDIM, HIDD, 1);
      gemm(bF, L.ff2, bY, ntok, HIDD, FFDIM, 0);
      lnorm(m, bY, L.ln2, m, ntok);
    }
    lnorm(m, nullptr, tp.en, m, ntok);
  };

  auto run_decoder = [&](const TrP& tp, half_t* x, const half_t* mem, int nseq,
                         int Lq, int Lk, int kvg) {
    const int ntok = nseq * Lq;
    const int memtok = (nseq / kvg) * Lk;
    for (int i = 0; i < tp.nd; ++i) {
      const DecL& L = tp.dec[i];
      gemm(x, L.sa.q, bQ, ntok, HIDD, HIDD, 0);
      gemm(x, L.sa.k, bK, ntok, HIDD, HIDD, 0);
      gemm(x, L.sa.v, bV, ntok, HIDD, HIDD, 0);
      attn(bQ, bK, bV, bA, Lq, Lq, 1, nseq);
      gemm(bA, L.sa.o, bY, ntok, HIDD, HIDD, 0);
      lnorm(x, bY, L.ln1, x, ntok);
      gemm(x, L.ca.q, bQ, ntok, HIDD, HIDD, 0);
      gemm(mem, L.ca.k, bK, memtok, HIDD, HIDD, 0);
      gemm(mem, L.ca.v, bV, memtok, HIDD, HIDD, 0);
      attn(bQ, bK, bV, bA, Lq, Lk, kvg, nseq);
      gemm(bA, L.ca.o, bY, ntok, HIDD, HIDD, 0);
      lnorm(x, bY, L.ln2, x, ntok);
      gemm(x, L.ff1, bF, ntok, FFDIM, HIDD, 1);
      gemm(bF, L.ff2, bY, ntok, HIDD, FFDIM, 0);
      lnorm(x, bY, L.ln3, x, ntok);
    }
    lnorm(x, nullptr, tp.dn, x, ntok);
  };

  // mixin decoder: Lq==1 so self-attn collapses to o(v(x))
  auto run_mixin_dec = [&](const TrP& tp, half_t* x, const half_t* mem,
                           int ntok, int Tl) {
    for (int i = 0; i < tp.nd; ++i) {
      const DecL& L = tp.dec[i];
      gemm(x, L.sa.v, bV, ntok, HIDD, HIDD, 0);
      gemm(bV, L.sa.o, bY, ntok, HIDD, HIDD, 0);
      lnorm(x, bY, L.ln1, x, ntok);
      gemm(x, L.ca.q, bQ, ntok, HIDD, HIDD, 0);
      gemm(mem, L.ca.k, bK, BATCH * SAMPLES, HIDD, HIDD, 0);
      gemm(mem, L.ca.v, bV, BATCH * SAMPLES, HIDD, HIDD, 0);
      attn(bQ, bK, bV, bA, 1, SAMPLES, Tl, ntok);
      gemm(bA, L.ca.o, bY, ntok, HIDD, HIDD, 0);
      lnorm(x, bY, L.ln2, x, ntok);
      gemm(x, L.ff1, bF, ntok, FFDIM, HIDD, 1);
      gemm(bF, L.ff2, bY, ntok, HIDD, FFDIM, 0);
      lnorm(x, bY, L.ln3, x, ntok);
    }
    lnorm(x, nullptr, tp.dn, x, ntok);
  };

  // ---- forward pass ----
  embed_kernel<<<dim3(NTOK / GBM, HIDD / GBN), 128, 0, stream>>>(
      dataF, embed.w, embed.b, emb16, emb32, NTOK);
  hipMemcpyAsync(X, emb16, (size_t)NTOK * HIDD * 2, hipMemcpyDeviceToDevice, stream);
  cvt_f32_f16_kernel<<<(BATCH * SSTATE * HIDD + 255) / 256, 256, 0, stream>>>(
      stateF, state16, BATCH * SSTATE * HIDD);

  // level 0: 512 seqs x 32 tokens
  run_encoder(levels[0], state16, memBuf, BATCH, SSTATE);
  run_decoder(levels[0], X, memBuf, 512, 32, SSTATE, 64);
  gather_stride_kernel<<<512, HIDD, 0, stream>>>(X, res0, 32);

  // level 1: same token buffer viewed as 256 seqs x 64 tokens
  run_encoder(levels[1], state16, memBuf, BATCH, SSTATE);
  run_decoder(levels[1], X, memBuf, 256, 64, SSTATE, 32);
  compact_half_kernel<<<dim3(256, 32), HIDD, 0, stream>>>(X, Xc);
  gather_stride_kernel<<<256, HIDD, 0, stream>>>(Xc, res1, 32);

  // level 2: Xc viewed as 128 seqs x 64 tokens
  run_encoder(levels[2], state16, memBuf, BATCH, SSTATE);
  run_decoder(levels[2], Xc, memBuf, 128, 64, SSTATE, 16);
  compact_half_kernel<<<dim3(128, 32), HIDD, 0, stream>>>(Xc, Xc2);
  gather_stride_kernel<<<128, HIDD, 0, stream>>>(Xc2, res2, 32);

  // top-level sample: logits[b*16+t][s], categorical over t (one-hot forward)
  gemm(res2, head, Lh, BATCH * 16, SAMPLES, HIDD, 0);
  sample_top_kernel<<<1, 256, 0, stream>>>(Lh, idx2, 16, SAMPLES, 3u);
  gather_idx_kernel<<<BATCH * SAMPLES, HIDD, 0, stream>>>(res2, idx2, Fdata,
                                                          SAMPLES, 16);

  // mixin pass, lev = 1 (Tl = 32)
  run_encoder(mixin, Fdata, memBuf, BATCH, SAMPLES);
  hipMemcpyAsync(Xm, res1, (size_t)256 * HIDD * 2, hipMemcpyDeviceToDevice, stream);
  run_mixin_dec(mixin, Xm, memBuf, 256, 32);
  gemm(Xm, hbl1[1], bQ, 256, HIDD, HIDD, 1);
  gemm(bQ, hbl2[1], Lh, 256, SAMPLES, HIDD, 0);
  sample_child_kernel<<<1, 256, 0, stream>>>(Lh, idx2, idx1, 32, SAMPLES, 1u);
  gather_idx_kernel<<<BATCH * SAMPLES, HIDD, 0, stream>>>(res1, idx1, Fd1,
                                                          SAMPLES, 32);
  concat_kernel<<<256, HIDD, 0, stream>>>(Fd1, Fdata, catB);
  gemm(catB, rnn1, bQ, 256, HIDD, 512, 1);
  gemm(bQ, rnn2, Fdata, 256, HIDD, HIDD, 0);

  // mixin pass, lev = 0 (Tl = 64); post-loop Fdata update is dead for output
  run_encoder(mixin, Fdata, memBuf, BATCH, SAMPLES);
  hipMemcpyAsync(Xm, res0, (size_t)512 * HIDD * 2, hipMemcpyDeviceToDevice, stream);
  run_mixin_dec(mixin, Xm, memBuf, 512, 64);
  gemm(Xm, hbl1[0], bQ, 512, HIDD, HIDD, 1);
  gemm(bQ, hbl2[0], Lh, 512, SAMPLES, HIDD, 0);
  sample_child_kernel<<<1, 256, 0, stream>>>(Lh, idx1, idx0, 64, SAMPLES, 0u);

  // final: counts, stable ranks of selected chunks, zero + scatter
  count_rank_kernel<<<BATCH, 64, 0, stream>>>(idx0, cnt, rank);
  zero_f32_kernel<<<(BATCH * OUTLEN * HIDD + 255) / 256, 256, 0, stream>>>(
      (float*)d_out, BATCH * OUTLEN * HIDD);
  scatter_out_kernel<<<dim3(64, BATCH), HIDD, 0, stream>>>(emb32, cnt, rank,
                                                           (float*)d_out);
}
